// CausalSelfAttention_67087389163583
// MI455X (gfx1250) — matrix operand order repaired and checked by reading, not compiled, not run on hardware
//
#include <hip/hip_runtime.h>

#define BATCH 2
#define SEQ   4096
#define EMB   768
#define NH    12
#define DH    64
#define BT    (BATCH*SEQ)   /* 8192 */
#define MT    (BT/16)       /* 512 */
#define MT4   (MT/4)        /* 128 */
#define NT    (EMB/16)      /* 48  */
#define WAVES 8

typedef __attribute__((ext_vector_type(16))) __bf16 v16bf;
typedef __attribute__((ext_vector_type(8)))  __bf16 v8bf;
typedef __attribute__((ext_vector_type(8)))  float  v8f;
typedef __attribute__((ext_vector_type(4)))  float  v4f;

#define WMMA_BF16(A, B, C) \
  __builtin_amdgcn_wmma_f32_16x16x32_bf16(false, (A), false, (B), (short)0, (C), false, false)

// ---------------------------------------------------------------------------
// Kernel 0: bulk f32 -> bf16 conversion (x and the four weight matrices).
// ---------------------------------------------------------------------------
__global__ __launch_bounds__(256) void to_bf16(const float* __restrict__ s,
                                               __bf16* __restrict__ d, int n8) {
  const int i = blockIdx.x * 256 + threadIdx.x;
  if (i >= n8) return;
  const v4f a = *(const v4f*)(s + (size_t)i * 8);
  const v4f b = *(const v4f*)(s + (size_t)i * 8 + 4);
  v8bf o;
  o[0] = (__bf16)a[0]; o[1] = (__bf16)a[1]; o[2] = (__bf16)a[2]; o[3] = (__bf16)a[3];
  o[4] = (__bf16)b[0]; o[5] = (__bf16)b[1]; o[6] = (__bf16)b[2]; o[7] = (__bf16)b[3];
  *(v8bf*)(d + (size_t)i * 8) = o;
}

// A-fragment (16x32 bf16): 16 contiguous k elements (k = 16*hi + s)
__device__ __forceinline__ v16bf load_afrag(const __bf16* __restrict__ p) {
  v8bf c0 = *(const v8bf*)p;
  v8bf c1 = *(const v8bf*)(p + 8);
  v16bf a;
#pragma unroll
  for (int e = 0; e < 8; ++e) { a[e] = c0[e]; a[e + 8] = c1[e]; }
  return a;
}

// Ping-pong GEMM inner-loop helpers (no register rotation -> no v_mov shuffles)
#define LOADF(buf, kk)                                   \
  b##buf  = *(const v16bf*)(wr + (kk));                  \
  a##buf##0 = load_afrag(xr0 + (kk));                    \
  a##buf##1 = load_afrag(xr1 + (kk));                    \
  a##buf##2 = load_afrag(xr2 + (kk));                    \
  a##buf##3 = load_afrag(xr3 + (kk));

#define COMP(buf)                                        \
  acc0 = WMMA_BF16(a##buf##0, b##buf, acc0);             \
  acc1 = WMMA_BF16(a##buf##1, b##buf, acc1);             \
  acc2 = WMMA_BF16(a##buf##2, b##buf, acc2);             \
  acc3 = WMMA_BF16(a##buf##3, b##buf, acc3);

// ---------------------------------------------------------------------------
// Kernel 1: fused Q/K/V projection, one wave = 64x16 slab, double-buffered.
// ---------------------------------------------------------------------------
__global__ __launch_bounds__(256) void qkv_proj(
    const __bf16* __restrict__ xb,
    const __bf16* __restrict__ wqb, const float* __restrict__ bq,
    const __bf16* __restrict__ wkb, const float* __restrict__ bk,
    const __bf16* __restrict__ wvb, const float* __restrict__ bv,
    __bf16* __restrict__ qo, __bf16* __restrict__ ko, __bf16* __restrict__ vo)
{
  const int wid  = __builtin_amdgcn_readfirstlane(blockIdx.x * WAVES + (threadIdx.x >> 5));
  const int lane = threadIdx.x & 31;
  const int mat  = wid / (MT4 * NT);       // 0=Q 1=K 2=V  (scalar)
  const int rem  = wid % (MT4 * NT);
  const int tm4  = rem / NT;
  const int tn   = rem % NT;
  const int row0 = tm4 * 64;

  const __bf16* w    = (mat == 0) ? wqb : (mat == 1) ? wkb : wvb;
  const float*  bias = (mat == 0) ? bq  : (mat == 1) ? bk  : bv;

  const int lm    = lane & 15;
  const int hi    = lane >> 4;
  const int koffA = hi ? 16 : 0;
  const int koffB = hi ? 16 : 0;

  const __bf16* wr  = w + (size_t)(tn * 16 + lm) * EMB + koffB;
  const __bf16* xr0 = xb + (size_t)(row0 +  0 + lm) * EMB + koffA;
  const __bf16* xr1 = xb + (size_t)(row0 + 16 + lm) * EMB + koffA;
  const __bf16* xr2 = xb + (size_t)(row0 + 32 + lm) * EMB + koffA;
  const __bf16* xr3 = xb + (size_t)(row0 + 48 + lm) * EMB + koffA;

  v8f acc0 = {0.f,0.f,0.f,0.f,0.f,0.f,0.f,0.f};
  v8f acc1 = acc0, acc2 = acc0, acc3 = acc0;

  v16bf bA, aA0, aA1, aA2, aA3;
  v16bf bB, aB0, aB1, aB2, aB3;

  LOADF(A, 0)                                   // prologue: step 0
  int kb = 0;
  for (int i = 0; i < (EMB / 64) - 1; ++i) {    // 11 iterations, 2 steps each
    LOADF(B, kb + 32)
    COMP(A)
    LOADF(A, kb + 64)
    COMP(B)
    kb += 64;
  }
  LOADF(B, kb + 32)                             // epilogue: steps 704, 736
  COMP(A)
  COMP(B)

  const int n = tn * 16 + lm;
  const int h = n >> 6, d = n & 63;
  const float bias_v = bias[n];

  if (mat == 2) {                               // V^T: packed 8x bf16 stores
    auto store_v = [&](const v8f& a, int s) {
      const int t0 = row0 + s * 16 + 8 * hi;
      const int bb = t0 >> 12, t = t0 & (SEQ - 1);
      v8bf pk;
#pragma unroll
      for (int r = 0; r < 8; ++r) pk[r] = (__bf16)(a[r] + bias_v);
      *(v8bf*)(vo + ((size_t)(bb * NH + h) * DH + d) * SEQ + t) = pk;
    };
    store_v(acc0, 0); store_v(acc1, 1); store_v(acc2, 2); store_v(acc3, 3);
  } else {
    __bf16* dst = (mat == 0) ? qo : ko;
    const float scale = (mat == 0) ? 0.125f : 1.0f;
    auto store_qk = [&](const v8f& a, int s) {
#pragma unroll
      for (int r = 0; r < 8; ++r) {
        const int mg = row0 + s * 16 + 8 * hi + r;
        const int bb = mg >> 12, t = mg & (SEQ - 1);
        dst[((size_t)(bb * NH + h) * SEQ + t) * DH + d] =
            (__bf16)((a[r] + bias_v) * scale);
      }
    };
    store_qk(acc0, 0); store_qk(acc1, 1); store_qk(acc2, 2); store_qk(acc3, 3);
  }
}

// ---------------------------------------------------------------------------
// Kernel 2: causal flash attention.  One wave = 16 query rows; 64-wide key
// blocks.  V fragments are loaded before the softmax VALU section so they
// overlap with the shuffle-reductions / exp / LDS transpose bounce.
// ---------------------------------------------------------------------------
__global__ __launch_bounds__(256) void flash_attn(
    const __bf16* __restrict__ q, const __bf16* __restrict__ k,
    const __bf16* __restrict__ vt, __bf16* __restrict__ attn)
{
  __shared__ __align__(16) __bf16 pbuf[WAVES][16][72];  // 144B row stride

  const int wv_   = threadIdx.x >> 5;
  const int lane  = threadIdx.x & 31;
  const int wid   = __builtin_amdgcn_readfirstlane(blockIdx.x * WAVES + wv_);
  const int bh    = wid >> 8;
  const int qbase = (wid & 255) * 16;
  const int b     = bh / NH, h = bh % NH;

  const int lm    = lane & 15;
  const int hi    = lane >> 4;
  const int koffA = hi ? 16 : 0;
  const int koffB = hi ? 16 : 0;

  const __bf16* qrow = q + ((size_t)bh * SEQ + qbase + lm) * DH;
  const v16bf qa0 = load_afrag(qrow + koffA);
  const v16bf qa1 = load_afrag(qrow + 32 + koffA);

  v8f o0 = {0.f,0.f,0.f,0.f,0.f,0.f,0.f,0.f};
  v8f o1 = o0, o2 = o0, o3 = o0;
  float mr[8], lr[8];
#pragma unroll
  for (int r = 0; r < 8; ++r) { mr[r] = -3.0e38f; lr[r] = 0.f; }

  const __bf16* kb_p = k  + (size_t)bh * SEQ * DH;
  const __bf16* vb_p = vt + (size_t)bh * DH * SEQ;

  for (int kb = 0; kb <= qbase; kb += 64) {
    // ---- S = (Q*scale) @ K^T : four 16x16 tiles covering keys kb..kb+63
    v8f s0 = {0.f,0.f,0.f,0.f,0.f,0.f,0.f,0.f};
    v8f s1 = s0, s2 = s0, s3 = s0;
    {
      const __bf16* kr = kb_p + (size_t)(kb + lm) * DH;
      s0 = WMMA_BF16(qa0, *(const v16bf*)(kr + koffB),      s0);
      s0 = WMMA_BF16(qa1, *(const v16bf*)(kr + 32 + koffB), s0);
      kr += 16 * DH;
      s1 = WMMA_BF16(qa0, *(const v16bf*)(kr + koffB),      s1);
      s1 = WMMA_BF16(qa1, *(const v16bf*)(kr + 32 + koffB), s1);
      kr += 16 * DH;
      s2 = WMMA_BF16(qa0, *(const v16bf*)(kr + koffB),      s2);
      s2 = WMMA_BF16(qa1, *(const v16bf*)(kr + 32 + koffB), s2);
      kr += 16 * DH;
      s3 = WMMA_BF16(qa0, *(const v16bf*)(kr + koffB),      s3);
      s3 = WMMA_BF16(qa1, *(const v16bf*)(kr + 32 + koffB), s3);
    }

    // ---- issue V fragment loads now; they overlap the softmax VALU below
    const __bf16* vr0 = vb_p + (size_t)( 0 + lm) * SEQ + kb + koffB;
    const __bf16* vr1 = vb_p + (size_t)(16 + lm) * SEQ + kb + koffB;
    const __bf16* vr2 = vb_p + (size_t)(32 + lm) * SEQ + kb + koffB;
    const __bf16* vr3 = vb_p + (size_t)(48 + lm) * SEQ + kb + koffB;
    const v16bf vf00 = *(const v16bf*)(vr0);
    const v16bf vf01 = *(const v16bf*)(vr0 + 32);
    const v16bf vf10 = *(const v16bf*)(vr1);
    const v16bf vf11 = *(const v16bf*)(vr1 + 32);
    const v16bf vf20 = *(const v16bf*)(vr2);
    const v16bf vf21 = *(const v16bf*)(vr2 + 32);
    const v16bf vf30 = *(const v16bf*)(vr3);
    const v16bf vf31 = *(const v16bf*)(vr3 + 32);

    if (kb + 64 <= qbase)
      __builtin_prefetch(kb_p + (size_t)(kb + 64 + lm) * DH, 0, 1);

    // ---- causal mask (D layout: row = qbase + r + 8*hi, col = kb+16t+lm)
    const int mg0 = qbase + 8 * hi;
    const int n0 = kb + lm;
#pragma unroll
    for (int r = 0; r < 8; ++r) {
      const int m = mg0 + r;
      if (n0      > m) s0[r] = -3.0e38f;
      if (n0 + 16 > m) s1[r] = -3.0e38f;
      if (n0 + 32 > m) s2[r] = -3.0e38f;
      if (n0 + 48 > m) s3[r] = -3.0e38f;
    }

    // ---- online softmax
#pragma unroll
    for (int r = 0; r < 8; ++r) {
      float t = fmaxf(fmaxf(s0[r], s1[r]), fmaxf(s2[r], s3[r]));
      t = fmaxf(t, __shfl_xor(t, 1, 32));
      t = fmaxf(t, __shfl_xor(t, 2, 32));
      t = fmaxf(t, __shfl_xor(t, 4, 32));
      t = fmaxf(t, __shfl_xor(t, 8, 32));
      const float nm   = fmaxf(mr[r], t);
      const float corr = __expf(mr[r] - nm);
      const float p0 = __expf(s0[r] - nm);
      const float p1 = __expf(s1[r] - nm);
      const float p2 = __expf(s2[r] - nm);
      const float p3 = __expf(s3[r] - nm);
      float rs = (p0 + p1) + (p2 + p3);
      rs += __shfl_xor(rs, 1, 32);
      rs += __shfl_xor(rs, 2, 32);
      rs += __shfl_xor(rs, 4, 32);
      rs += __shfl_xor(rs, 8, 32);
      lr[r] = lr[r] * corr + rs;
      mr[r] = nm;
      o0[r] *= corr; o1[r] *= corr; o2[r] *= corr; o3[r] *= corr;
      __bf16* prow = &pbuf[wv_][8 * hi + r][lm];
      prow[0]  = (__bf16)p0;
      prow[16] = (__bf16)p1;
      prow[32] = (__bf16)p2;
      prow[48] = (__bf16)p3;
    }
    asm volatile("s_wait_dscnt 0" ::: "memory");

    // ---- P back as two A-fragments (keys kb..+31 and kb+32..+63)
    const v16bf pa0 = load_afrag(&pbuf[wv_][lm][koffA]);
    const v16bf pa1 = load_afrag(&pbuf[wv_][lm][32 + koffA]);

    // ---- O += P @ V with preloaded fragments
    o0 = WMMA_BF16(pa0, vf00, o0);
    o0 = WMMA_BF16(pa1, vf01, o0);
    o1 = WMMA_BF16(pa0, vf10, o1);
    o1 = WMMA_BF16(pa1, vf11, o1);
    o2 = WMMA_BF16(pa0, vf20, o2);
    o2 = WMMA_BF16(pa1, vf21, o2);
    o3 = WMMA_BF16(pa0, vf30, o3);
    o3 = WMMA_BF16(pa1, vf31, o3);
  }

  // ---- normalize and store merged heads [B,T,C] bf16
#pragma unroll
  for (int r = 0; r < 8; ++r) {
    const float inv = 1.0f / lr[r];
    const int m = qbase + r + 8 * hi;
    const size_t row = (size_t)(b * SEQ + m) * EMB + h * DH;
    attn[row +  0 + lm] = (__bf16)(o0[r] * inv);
    attn[row + 16 + lm] = (__bf16)(o1[r] * inv);
    attn[row + 32 + lm] = (__bf16)(o2[r] * inv);
    attn[row + 48 + lm] = (__bf16)(o3[r] * inv);
  }
}

// ---------------------------------------------------------------------------
// Kernel 3: output projection (bf16 x bf16 -> f32), double-buffered.
// ---------------------------------------------------------------------------
__global__ __launch_bounds__(256) void out_proj(
    const __bf16* __restrict__ attn, const __bf16* __restrict__ wob,
    float* __restrict__ out)
{
  const int wid  = __builtin_amdgcn_readfirstlane(blockIdx.x * WAVES + (threadIdx.x >> 5));
  const int lane = threadIdx.x & 31;
  const int tm4 = wid / NT, tn = wid % NT;
  const int row0 = tm4 * 64;
  const int lm = lane & 15, hi = lane >> 4;
  const int koffA = hi ? 16 : 0, koffB = hi ? 16 : 0;

  const __bf16* wr  = wob + (size_t)(tn * 16 + lm) * EMB + koffB;
  const __bf16* xr0 = attn + (size_t)(row0 +  0 + lm) * EMB + koffA;
  const __bf16* xr1 = attn + (size_t)(row0 + 16 + lm) * EMB + koffA;
  const __bf16* xr2 = attn + (size_t)(row0 + 32 + lm) * EMB + koffA;
  const __bf16* xr3 = attn + (size_t)(row0 + 48 + lm) * EMB + koffA;

  v8f acc0 = {0.f,0.f,0.f,0.f,0.f,0.f,0.f,0.f};
  v8f acc1 = acc0, acc2 = acc0, acc3 = acc0;

  v16bf bA, aA0, aA1, aA2, aA3;
  v16bf bB, aB0, aB1, aB2, aB3;

  LOADF(A, 0)
  int kb = 0;
  for (int i = 0; i < (EMB / 64) - 1; ++i) {
    LOADF(B, kb + 32)
    COMP(A)
    LOADF(A, kb + 64)
    COMP(B)
    kb += 64;
  }
  LOADF(B, kb + 32)
  COMP(A)
  COMP(B)

  auto store_o = [&](const v8f& a, int s) {
#pragma unroll
    for (int r = 0; r < 8; ++r) {
      const int mg = row0 + s * 16 + 8 * hi + r;
      out[(size_t)mg * EMB + tn * 16 + lm] = a[r];
    }
  };
  store_o(acc0, 0); store_o(acc1, 1); store_o(acc2, 2); store_o(acc3, 3);
}

// ---------------------------------------------------------------------------
extern "C" void kernel_launch(void* const* d_in, const int* in_sizes, int n_in,
                              void* d_out, int out_size, void* d_ws, size_t ws_size,
                              hipStream_t stream) {
  const float* x  = (const float*)d_in[0];
  const float* wq = (const float*)d_in[1];
  const float* bq = (const float*)d_in[2];
  const float* wk = (const float*)d_in[3];
  const float* bk = (const float*)d_in[4];
  const float* wv = (const float*)d_in[5];
  const float* bv = (const float*)d_in[6];
  const float* wo = (const float*)d_in[7];
  float* out = (float*)d_out;

  const size_t HEADBUF = (size_t)BATCH * NH * SEQ * DH * sizeof(__bf16); // 12.6MB
  const size_t WBUF    = (size_t)EMB * EMB * sizeof(__bf16);             // 1.18MB
  char* ws = (char*)d_ws;
  __bf16* xbw = (__bf16*)(ws);
  __bf16* qw  = (__bf16*)(ws + 1 * HEADBUF);
  __bf16* kw  = (__bf16*)(ws + 2 * HEADBUF);
  __bf16* vw  = (__bf16*)(ws + 3 * HEADBUF);
  __bf16* aw  = (__bf16*)(ws + 4 * HEADBUF);
  __bf16* wqb = (__bf16*)(ws + 5 * HEADBUF);
  __bf16* wkb = (__bf16*)(ws + 5 * HEADBUF + 1 * WBUF);
  __bf16* wvb = (__bf16*)(ws + 5 * HEADBUF + 2 * WBUF);
  __bf16* wob = (__bf16*)(ws + 5 * HEADBUF + 3 * WBUF);

  dim3 blk(256);
  const int NX8 = BT * EMB / 8;     // 786432
  const int NW8 = EMB * EMB / 8;    // 73728
  to_bf16<<<(NX8 + 255) / 256, blk, 0, stream>>>(x,  xbw, NX8);
  to_bf16<<<(NW8 + 255) / 256, blk, 0, stream>>>(wq, wqb, NW8);
  to_bf16<<<(NW8 + 255) / 256, blk, 0, stream>>>(wk, wkb, NW8);
  to_bf16<<<(NW8 + 255) / 256, blk, 0, stream>>>(wv, wvb, NW8);
  to_bf16<<<(NW8 + 255) / 256, blk, 0, stream>>>(wo, wob, NW8);

  qkv_proj  <<<(3 * MT4 * NT) / WAVES,             blk, 0, stream>>>(
      xbw, wqb, bq, wkb, bk, wvb, bv, qw, kw, vw);
  flash_attn<<<(BATCH * NH * (SEQ / 16)) / WAVES,  blk, 0, stream>>>(qw, kw, vw, aw);
  out_proj  <<<(MT4 * NT) / WAVES,                 blk, 0, stream>>>(aw, wob, out);
}
